// DSSM_17051020165813
// MI455X (gfx1250) — matrix operand order repaired and checked by reading, not compiled, not run on hardware
//
#include <hip/hip_runtime.h>
#include <hip/hip_bf16.h>
#include <math.h>

// Problem sizes (match reference)
#define B_   128
#define T_   512
#define F_   128
#define H_   512
#define S_   64
#define TH_  256
#define EH_  256
#define THREEH 1536
#define KINF 1088              // 2H + S
#define LOG2PI_ 1.8378770664093453f

typedef __attribute__((ext_vector_type(16))) _Float16 v16h;
typedef __attribute__((ext_vector_type(8)))  float    v8f;
typedef _Float16 h16;

// ---------------------------------------------------------------------------
// CDNA5 async global->LDS copy (ASYNCcnt path).  Per CDNA5 ISA:
//   each lane copies 16 bytes from its global address to its LDS byte offset
// VDST holds the LDS byte offset (low 32 bits of the generic shared pointer),
// VADDR holds the 64-bit global address.  16B alignment on both sides.
// ---------------------------------------------------------------------------
__device__ __forceinline__ void async_b128(unsigned int lds_off, const void* g) {
  asm volatile("global_load_async_to_lds_b128 %0, %1, off"
               :: "v"(lds_off), "v"((unsigned long long)(uintptr_t)g)
               : "memory");
}
__device__ __forceinline__ void wait_async0() {
  asm volatile("s_wait_asynccnt 0x0" ::: "memory");
}

// ---------------------------------------------------------------------------
// Fragment helpers.  CDNA5 wave32 WMMA 16x16x32 f16:
//  A (16x32): lanes 0-15 = M rows (lanes 16-31 repeat M with K+8 offset);
//   per lane 16 halves: j in [0,8) -> k = kb + half*8 + j, j in [8,16) -> +16.
//  B is consumed from a pre-packed layout where each lane's 16 halves are
//  contiguous (one 32-byte load), produced by pack_b below.
// ---------------------------------------------------------------------------
__device__ __forceinline__ v16h lds_a_frag(const h16* base, int rowStrideH,
                                           int lane, int kbase) {
  const int m = lane & 15, half = lane >> 4;
  const h16* p0 = base + m * rowStrideH + kbase + half * 8;
  v16h a;
  *((uint4*)&a)     = *(const uint4*)p0;        // halves 0..7
  *(((uint4*)&a)+1) = *(const uint4*)(p0 + 16); // halves 8..15
  return a;
}

__device__ __forceinline__ v16h ld_b_frag(const h16* __restrict__ P, int KT,
                                          int nt, int kt, int lane) {
  const h16* p = P + (((size_t)nt * KT + kt) * 512) + (size_t)lane * 16;
  v16h b;
  *((uint4*)&b)     = *(const uint4*)p;
  *(((uint4*)&b)+1) = *(const uint4*)(p + 8);
  return b;
}

__device__ __forceinline__ float sigmoidf_(float x) {
  return 1.0f / (1.0f + expf(-x));
}

// ---------------------------------------------------------------------------
// Pack fp32 weight W[N][K] (gemm computes act @ W^T) into f16 WMMA B-fragment
// order: linear index = ((nt*KT + kt)*32 + lane)*16 + j
//   n = nt*16 + (lane&15);  k = kt*32 + (j>>3)*16 + (lane>>4)*8 + (j&7)
// ---------------------------------------------------------------------------
__global__ void pack_b(const float* __restrict__ W, h16* __restrict__ P,
                       int N, int K) {
  const int KT = K >> 5;
  const long total = (long)N * K;
  for (long idx = (long)blockIdx.x * blockDim.x + threadIdx.x; idx < total;
       idx += (long)gridDim.x * blockDim.x) {
    long e = idx;
    const int j    = (int)(e & 15); e >>= 4;
    const int lane = (int)(e & 31); e >>= 5;
    const int kt   = (int)(e % KT);
    const int nt   = (int)(e / KT);
    const int half = lane >> 4;
    const int n = nt * 16 + (lane & 15);
    const int k = kt * 32 + ((j >> 3) << 4) + half * 8 + (j & 7);
    P[idx] = (h16)W[(long)n * K + k];
  }
}

__global__ void cvt_f16(const float* __restrict__ src, h16* __restrict__ dst,
                        long n) {
  for (long i = (long)blockIdx.x * blockDim.x + threadIdx.x; i < n;
       i += (long)gridDim.x * blockDim.x) dst[i] = (h16)src[i];
}

__global__ void zero_f32(float* p, int n) {
  for (int i = blockIdx.x * blockDim.x + threadIdx.x; i < n;
       i += gridDim.x * blockDim.x) p[i] = 0.0f;
}

__global__ void zero_f16(h16* p, int n) {
  for (int i = blockIdx.x * blockDim.x + threadIdx.x; i < n;
       i += gridDim.x * blockDim.x) p[i] = (h16)0.0f;
}

// ---------------------------------------------------------------------------
// One bidirectional-encoder GRU step (blockIdx.y = direction).
// 32 WGs x 8 waves; WG owns M-tile mi (16 batch rows) and N-group ng
// (8 waves x 16 hidden cols).  Each wave: 6 accumulators, 60 WMMAs.
// All A staging is an async f16 copy into LDS.
// ---------------------------------------------------------------------------
__global__ void __launch_bounds__(256)
enc_step(const h16* __restrict__ x16,
         const h16* Pih_f, const h16* Phh_f,
         const float* bih_f, const float* bhh_f,
         const h16* Pih_b, const h16* Phh_b,
         const float* bih_b, const float* bhh_b,
         const float* hprev_f, float* hnext_f,
         const h16* hprev16_f, h16* hnext16_f,
         const float* hprev_b, float* hnext_b,
         const h16* hprev16_b, h16* hnext16_b,
         h16* __restrict__ hb16, int s) {
  const int dir = blockIdx.y;
  const int t   = dir ? (T_ - 1 - s) : s;
  const h16*   Pih     = dir ? Pih_b     : Pih_f;
  const h16*   Phh     = dir ? Phh_b     : Phh_f;
  const float* bih     = dir ? bih_b     : bih_f;
  const float* bhh     = dir ? bhh_b     : bhh_f;
  const float* hprev   = dir ? hprev_b   : hprev_f;
  const h16*   hprev16 = dir ? hprev16_b : hprev16_f;
  float*       hnext   = dir ? hnext_b   : hnext_f;
  h16*         hnext16 = dir ? hnext16_b : hnext16_f;

  __shared__ __align__(16) h16 Ax[16 * F_];   //  4 KB
  __shared__ __align__(16) h16 Ah[16 * H_];   // 16 KB

  const int tid = threadIdx.x;
  const int mi  = blockIdx.x & 7;
  const int ng  = blockIdx.x >> 3;

  // Async-stage A tiles (pure f16 copies; ASYNCcnt tracked).
  const unsigned int axb = (unsigned int)(uintptr_t)Ax;
  const unsigned int ahb = (unsigned int)(uintptr_t)Ah;
  {
    const int r = tid >> 4, seg = tid & 15;       // 16 rows x 16 segs of 16B
    const h16* g = x16 + (size_t)(mi * 16 + r) * T_ * F_ + (size_t)t * F_ + seg * 8;
    async_b128(axb + (unsigned int)(r * (F_ * 2) + seg * 16), g);
  }
#pragma unroll
  for (int j = 0; j < 4; ++j) {                   // 16 rows x 64 segs of 16B
    const int i = j * 256 + tid;
    const int r = i >> 6, seg = i & 63;
    const h16* g = hprev16 + (size_t)(mi * 16 + r) * H_ + seg * 8;
    async_b128(ahb + (unsigned int)(r * (H_ * 2) + seg * 16), g);
  }
  wait_async0();
  __syncthreads();

  const int wv = tid >> 5, lane = tid & 31;
  const int nj  = ng * 8 + wv;            // hidden tile 0..31
  const int ntr = nj, ntz = nj + 32, ntn = nj + 64;

  v8f axr = {}, axz = {}, axn = {}, ahr = {}, ahz = {}, ahn = {};

#pragma unroll
  for (int kt = 0; kt < 4; ++kt) {        // x @ wih^T, K = 128
    v16h a  = lds_a_frag(Ax, F_, lane, kt * 32);
    v16h br = ld_b_frag(Pih, 4, ntr, kt, lane);
    v16h bz = ld_b_frag(Pih, 4, ntz, kt, lane);
    v16h bn = ld_b_frag(Pih, 4, ntn, kt, lane);
    axr = __builtin_amdgcn_wmma_f32_16x16x32_f16(false, a, false, br, (short)0, axr, false, false);
    axz = __builtin_amdgcn_wmma_f32_16x16x32_f16(false, a, false, bz, (short)0, axz, false, false);
    axn = __builtin_amdgcn_wmma_f32_16x16x32_f16(false, a, false, bn, (short)0, axn, false, false);
  }
#pragma unroll 2
  for (int kt = 0; kt < 16; ++kt) {       // h @ whh^T, K = 512
    v16h a  = lds_a_frag(Ah, H_, lane, kt * 32);
    v16h br = ld_b_frag(Phh, 16, ntr, kt, lane);
    v16h bz = ld_b_frag(Phh, 16, ntz, kt, lane);
    v16h bn = ld_b_frag(Phh, 16, ntn, kt, lane);
    ahr = __builtin_amdgcn_wmma_f32_16x16x32_f16(false, a, false, br, (short)0, ahr, false, false);
    ahz = __builtin_amdgcn_wmma_f32_16x16x32_f16(false, a, false, bz, (short)0, ahz, false, false);
    ahn = __builtin_amdgcn_wmma_f32_16x16x32_f16(false, a, false, bn, (short)0, ahn, false, false);
  }

  // Fused GRU gates; C/D layout: elem (m,n): n = lane&15, m = v + 8*(lane>>4)
  const int half = lane >> 4, nl = lane & 15;
  const int col  = nj * 16 + nl;
  const float bir = bih[col],          bhr = bhh[col];
  const float biz = bih[H_ + col],     bhz = bhh[H_ + col];
  const float bin_ = bih[2 * H_ + col], bhn = bhh[2 * H_ + col];
#pragma unroll
  for (int v = 0; v < 8; ++v) {
    const int row = mi * 16 + v + 8 * half;
    const float hp = hprev[row * H_ + col];
    const float r  = sigmoidf_(axr[v] + ahr[v] + bir + bhr);
    const float z  = sigmoidf_(axz[v] + ahz[v] + biz + bhz);
    const float n  = tanhf(axn[v] + bin_ + r * (ahn[v] + bhn));
    const float h  = (1.0f - z) * n + z * hp;
    hnext[row * H_ + col]   = h;
    hnext16[row * H_ + col] = (h16)h;
    hb16[(size_t)t * B_ * (2 * H_) + (size_t)row * (2 * H_) + dir * H_ + col] = (h16)h;
  }
}

// ---------------------------------------------------------------------------
// Inference GRU step: input = [h_birnn_t (1024) | prev_c (64)], K = 1088,
// plus recurrent K = 512.  150 WMMAs per 16x16 tile.
// ---------------------------------------------------------------------------
__global__ void __launch_bounds__(256)
inf_step(const h16* __restrict__ hb16, const float* __restrict__ cprev,
         const float* __restrict__ hprev, float* __restrict__ hnext,
         const h16* __restrict__ hprev16, h16* __restrict__ hnext16,
         const h16* __restrict__ Pih, const h16* __restrict__ Phh,
         const float* __restrict__ bih, const float* __restrict__ bhh, int t) {
  __shared__ __align__(16) h16 Ai[16 * KINF]; // 34 KB
  __shared__ __align__(16) h16 Ah[16 * H_];   // 16 KB

  const int tid = threadIdx.x;
  const int mi  = blockIdx.x & 7;
  const int ng  = blockIdx.x >> 3;

  const unsigned int aib = (unsigned int)(uintptr_t)Ai;
  const unsigned int ahb = (unsigned int)(uintptr_t)Ah;
  // h_birnn part of A: 16 rows x 128 segs of 16B, async copy.
#pragma unroll
  for (int j = 0; j < 8; ++j) {
    const int i = j * 256 + tid;
    const int r = i >> 7, seg = i & 127;
    const h16* g = hb16 + (size_t)t * B_ * (2 * H_) +
                   (size_t)(mi * 16 + r) * (2 * H_) + seg * 8;
    async_b128(aib + (unsigned int)(r * (KINF * 2) + seg * 16), g);
  }
  // recurrent A: 16 rows x 64 segs of 16B, async copy.
#pragma unroll
  for (int j = 0; j < 4; ++j) {
    const int i = j * 256 + tid;
    const int r = i >> 6, seg = i & 63;
    const h16* g = hprev16 + (size_t)(mi * 16 + r) * H_ + seg * 8;
    async_b128(ahb + (unsigned int)(r * (H_ * 2) + seg * 16), g);
  }
  // prev_c tail (f32 -> f16 convert): 16 rows x 64 cols.
  for (int i = tid; i < 16 * S_; i += 256) {
    const int r = i >> 6, k = i & 63;
    Ai[r * KINF + 2 * H_ + k] = (h16)cprev[(mi * 16 + r) * S_ + k];
  }
  wait_async0();
  __syncthreads();

  const int wv = tid >> 5, lane = tid & 31;
  const int nj  = ng * 8 + wv;
  const int ntr = nj, ntz = nj + 32, ntn = nj + 64;

  v8f axr = {}, axz = {}, axn = {}, ahr = {}, ahz = {}, ahn = {};

#pragma unroll 2
  for (int kt = 0; kt < 34; ++kt) {       // inp @ inf_wih^T, K = 1088
    v16h a  = lds_a_frag(Ai, KINF, lane, kt * 32);
    v16h br = ld_b_frag(Pih, 34, ntr, kt, lane);
    v16h bz = ld_b_frag(Pih, 34, ntz, kt, lane);
    v16h bn = ld_b_frag(Pih, 34, ntn, kt, lane);
    axr = __builtin_amdgcn_wmma_f32_16x16x32_f16(false, a, false, br, (short)0, axr, false, false);
    axz = __builtin_amdgcn_wmma_f32_16x16x32_f16(false, a, false, bz, (short)0, axz, false, false);
    axn = __builtin_amdgcn_wmma_f32_16x16x32_f16(false, a, false, bn, (short)0, axn, false, false);
  }
#pragma unroll 2
  for (int kt = 0; kt < 16; ++kt) {       // h @ inf_whh^T, K = 512
    v16h a  = lds_a_frag(Ah, H_, lane, kt * 32);
    v16h br = ld_b_frag(Phh, 16, ntr, kt, lane);
    v16h bz = ld_b_frag(Phh, 16, ntz, kt, lane);
    v16h bn = ld_b_frag(Phh, 16, ntn, kt, lane);
    ahr = __builtin_amdgcn_wmma_f32_16x16x32_f16(false, a, false, br, (short)0, ahr, false, false);
    ahz = __builtin_amdgcn_wmma_f32_16x16x32_f16(false, a, false, bz, (short)0, ahz, false, false);
    ahn = __builtin_amdgcn_wmma_f32_16x16x32_f16(false, a, false, bn, (short)0, ahn, false, false);
  }

  const int half = lane >> 4, nl = lane & 15;
  const int col  = nj * 16 + nl;
  const float bir = bih[col],          bhr = bhh[col];
  const float biz = bih[H_ + col],     bhz = bhh[H_ + col];
  const float bin_ = bih[2 * H_ + col], bhn = bhh[2 * H_ + col];
#pragma unroll
  for (int v = 0; v < 8; ++v) {
    const int row = mi * 16 + v + 8 * half;
    const float hp = hprev[row * H_ + col];
    const float r  = sigmoidf_(axr[v] + ahr[v] + bir + bhr);
    const float z  = sigmoidf_(axz[v] + ahz[v] + biz + bhz);
    const float n  = tanhf(axn[v] + bin_ + r * (ahn[v] + bhn));
    const float h  = (1.0f - z) * n + z * hp;
    hnext[row * H_ + col]   = h;
    hnext16[row * H_ + col] = (h16)h;
  }
}

// ---------------------------------------------------------------------------
// Per-step heads: fc -> log_softmax, gumbel-softmax sample, transition prior,
// emission + recon, KL.  One block per batch row; deterministic reductions.
// ---------------------------------------------------------------------------
__global__ void __launch_bounds__(256)
inf_small(const float* __restrict__ hnew,
          const float* __restrict__ cprev, float* __restrict__ cnext,
          const float* __restrict__ x, const float* __restrict__ gum,
          const float* __restrict__ tau_p,
          const float* __restrict__ fc_w, const float* __restrict__ fc_b,
          const float* __restrict__ tr_w1, const float* __restrict__ tr_b1,
          const float* __restrict__ tr_w2, const float* __restrict__ tr_b2,
          const float* __restrict__ em_w1, const float* __restrict__ em_b1,
          const float* __restrict__ em_w2, const float* __restrict__ em_b2,
          float* __restrict__ kl_tb, float* __restrict__ recon_tb,
          float* __restrict__ qout, int t) {
  __shared__ float sh_h[H_];
  __shared__ float sh_cprev[S_];
  __shared__ float sh_v[S_];      // logits scratch
  __shared__ float sh_qlp[S_];
  __shared__ float sh_qp[S_];
  __shared__ float sh_cs[S_];
  __shared__ float sh_plp[S_];
  __shared__ float sh_hid[256];
  __shared__ float sh_red[256];
  __shared__ float s_a, s_b;

  const int b = blockIdx.x, tid = threadIdx.x;

  for (int i = tid; i < H_; i += 256) sh_h[i] = hnew[b * H_ + i];
  if (tid < S_) sh_cprev[tid] = cprev[b * S_ + tid];
  __syncthreads();

  // q_logits = h @ fc_w^T + fc_b
  if (tid < S_) {
    float s = fc_b[tid];
    const float* w = fc_w + tid * H_;
    for (int k = 0; k < H_; ++k) s += sh_h[k] * w[k];
    sh_v[tid] = s;
  }
  __syncthreads();
  if (tid == 0) {                          // logsumexp
    float mx = sh_v[0];
    for (int j = 1; j < S_; ++j) mx = fmaxf(mx, sh_v[j]);
    float su = 0.0f;
    for (int j = 0; j < S_; ++j) su += expf(sh_v[j] - mx);
    s_a = mx + logf(su);
  }
  __syncthreads();
  if (tid < S_) {
    const float lp = sh_v[tid] - s_a;
    sh_qlp[tid] = lp;
    const float p = expf(lp);
    sh_qp[tid] = p;
    qout[(size_t)b * T_ * S_ + (size_t)t * S_ + tid] = p;  // q_probs [B,T,S]
  }

  // gumbel-softmax soft sample
  const float tau = *tau_p;
  if (tid < S_)
    sh_red[tid] = (sh_v[tid] + gum[((size_t)t * B_ + b) * S_ + tid]) / tau;
  __syncthreads();
  if (tid == 0) {
    float mx = sh_red[0];
    for (int j = 1; j < S_; ++j) mx = fmaxf(mx, sh_red[j]);
    float su = 0.0f;
    for (int j = 0; j < S_; ++j) su += expf(sh_red[j] - mx);
    s_a = mx; s_b = su;
  }
  __syncthreads();
  if (tid < S_) {
    const float cs = expf(sh_red[tid] - s_a) / s_b;
    sh_cs[tid] = cs;
    cnext[b * S_ + tid] = cs;
  }
  __syncthreads();

  // prior log-probs
  if (t == 0) {
    if (tid < S_) sh_plp[tid] = -logf((float)S_);
    __syncthreads();
  } else {
    if (tid < TH_) {
      float s = tr_b1[tid];
      const float* w = tr_w1 + tid * S_;
      for (int k = 0; k < S_; ++k) s += sh_cprev[k] * w[k];
      sh_hid[tid] = fmaxf(s, 0.0f);
    }
    __syncthreads();
    if (tid < S_) {
      float s = tr_b2[tid];
      const float* w = tr_w2 + tid * TH_;
      for (int k = 0; k < TH_; ++k) s += sh_hid[k] * w[k];
      sh_v[tid] = s;
    }
    __syncthreads();
    if (tid == 0) {
      float mx = sh_v[0];
      for (int j = 1; j < S_; ++j) mx = fmaxf(mx, sh_v[j]);
      float su = 0.0f;
      for (int j = 0; j < S_; ++j) su += expf(sh_v[j] - mx);
      s_a = mx + logf(su);
    }
    __syncthreads();
    if (tid < S_) sh_plp[tid] = sh_v[tid] - s_a;
    __syncthreads();
  }

  if (tid == 0) {
    float kl = 0.0f;
    for (int j = 0; j < S_; ++j) kl += sh_qp[j] * (sh_qlp[j] - sh_plp[j]);
    kl_tb[t * B_ + b] = kl;
  }
  __syncthreads();

  // emission mu + recon contribution
  if (tid < EH_) {
    float s = em_b1[tid];
    const float* w = em_w1 + tid * S_;
    for (int k = 0; k < S_; ++k) s += sh_cs[k] * w[k];
    sh_hid[tid] = fmaxf(s, 0.0f);
  }
  __syncthreads();
  if (tid < F_) {
    float mu = em_b2[tid];
    const float* w = em_w2 + tid * EH_;
    for (int k = 0; k < EH_; ++k) mu += sh_hid[k] * w[k];
    const float d = x[(size_t)b * T_ * F_ + (size_t)t * F_ + tid] - mu;
    sh_red[tid] = 0.5f * d * d;
  } else {
    sh_red[tid] = 0.0f;
  }
  __syncthreads();
  if (tid == 0) {
    float r = 0.0f;
    for (int j = 0; j < F_; ++j) r += sh_red[j];
    recon_tb[t * B_ + b] = r + 0.5f * (float)F_ * LOG2PI_;
  }
}

__global__ void final_reduce(const float* __restrict__ kl_tb,
                             const float* __restrict__ recon_tb,
                             float* __restrict__ out) {
  __shared__ float red[256];
  const int tid = threadIdx.x;
  float k = 0.0f, r = 0.0f;
  for (int i = tid; i < T_ * B_; i += 256) { k += kl_tb[i]; r += recon_tb[i]; }
  red[tid] = k; __syncthreads();
  for (int o = 128; o > 0; o >>= 1) { if (tid < o) red[tid] += red[tid + o]; __syncthreads(); }
  const float ksum = red[0]; __syncthreads();
  red[tid] = r; __syncthreads();
  for (int o = 128; o > 0; o >>= 1) { if (tid < o) red[tid] += red[tid + o]; __syncthreads(); }
  const float rsum = red[0];
  if (tid == 0) {
    const float kl = ksum / (float)B_;
    const float rc = rsum / (float)B_;
    out[0] = rc + kl;   // elbo
    out[1] = rc;        // reconstruction_loss
    out[2] = kl;        // kl_loss
  }
}

// ---------------------------------------------------------------------------
extern "C" void kernel_launch(void* const* d_in, const int* in_sizes, int n_in,
                              void* d_out, int out_size, void* d_ws, size_t ws_size,
                              hipStream_t stream) {
  (void)in_sizes; (void)n_in; (void)out_size; (void)ws_size;
  const float* x       = (const float*)d_in[0];
  const float* tau     = (const float*)d_in[1];
  const float* gum     = (const float*)d_in[2];
  const float* bf_wih  = (const float*)d_in[3];
  const float* bf_whh  = (const float*)d_in[4];
  const float* bf_bih  = (const float*)d_in[5];
  const float* bf_bhh  = (const float*)d_in[6];
  const float* bb_wih  = (const float*)d_in[7];
  const float* bb_whh  = (const float*)d_in[8];
  const float* bb_bih  = (const float*)d_in[9];
  const float* bb_bhh  = (const float*)d_in[10];
  const float* in_wih  = (const float*)d_in[11];
  const float* in_whh  = (const float*)d_in[12];
  const float* in_bih  = (const float*)d_in[13];
  const float* in_bhh  = (const float*)d_in[14];
  const float* fc_w    = (const float*)d_in[15];
  const float* fc_b    = (const float*)d_in[16];
  const float* tr_w1   = (const float*)d_in[17];
  const float* tr_b1   = (const float*)d_in[18];
  const float* tr_w2   = (const float*)d_in[19];
  const float* tr_b2   = (const float*)d_in[20];
  const float* em_w1   = (const float*)d_in[21];
  const float* em_b1   = (const float*)d_in[22];
  const float* em_w2   = (const float*)d_in[23];
  const float* em_b2   = (const float*)d_in[24];
  float* out = (float*)d_out;

  // Carve workspace (256B aligned).
  char* p = (char*)d_ws;
  auto carve = [&](size_t bytes) -> void* {
    void* r = (void*)p;
    p += (bytes + 255) & ~(size_t)255;
    return r;
  };
  h16* Pbf_ih = (h16*)carve((size_t)THREEH * F_ * sizeof(h16));
  h16* Pbf_hh = (h16*)carve((size_t)THREEH * H_ * sizeof(h16));
  h16* Pbb_ih = (h16*)carve((size_t)THREEH * F_ * sizeof(h16));
  h16* Pbb_hh = (h16*)carve((size_t)THREEH * H_ * sizeof(h16));
  h16* Pin_ih = (h16*)carve((size_t)THREEH * KINF * sizeof(h16));
  h16* Pin_hh = (h16*)carve((size_t)THREEH * H_ * sizeof(h16));
  h16* hb16   = (h16*)carve((size_t)T_ * B_ * 2 * H_ * sizeof(h16)); // 134 MB
  h16* x16    = (h16*)carve((size_t)B_ * T_ * F_ * sizeof(h16));     //  17 MB
  float* henc[2][2];
  h16*   henc16[2][2];
  for (int d = 0; d < 2; ++d)
    for (int i = 0; i < 2; ++i) {
      henc[d][i]   = (float*)carve((size_t)B_ * H_ * sizeof(float));
      henc16[d][i] = (h16*)carve((size_t)B_ * H_ * sizeof(h16));
    }
  float* hinf[2];
  h16*   hinf16[2];
  for (int i = 0; i < 2; ++i) {
    hinf[i]   = (float*)carve((size_t)B_ * H_ * sizeof(float));
    hinf16[i] = (h16*)carve((size_t)B_ * H_ * sizeof(h16));
  }
  float* cbuf[2];
  for (int i = 0; i < 2; ++i) cbuf[i] = (float*)carve((size_t)B_ * S_ * sizeof(float));
  float* kl_tb = (float*)carve((size_t)T_ * B_ * sizeof(float));
  float* rc_tb = (float*)carve((size_t)T_ * B_ * sizeof(float));

  // Pack weights fp32 -> f16 WMMA-fragment layout; convert x once.
  pack_b<<<1024, 256, 0, stream>>>(bf_wih, Pbf_ih, THREEH, F_);
  pack_b<<<1024, 256, 0, stream>>>(bf_whh, Pbf_hh, THREEH, H_);
  pack_b<<<1024, 256, 0, stream>>>(bb_wih, Pbb_ih, THREEH, F_);
  pack_b<<<1024, 256, 0, stream>>>(bb_whh, Pbb_hh, THREEH, H_);
  pack_b<<<2048, 256, 0, stream>>>(in_wih, Pin_ih, THREEH, KINF);
  pack_b<<<1024, 256, 0, stream>>>(in_whh, Pin_hh, THREEH, H_);
  cvt_f16<<<2048, 256, 0, stream>>>(x, x16, (long)B_ * T_ * F_);

  // Zero initial states (h0 = 0, c0 = 0), f32 + f16 mirrors.
  zero_f32<<<64, 256, 0, stream>>>(henc[0][0], B_ * H_);
  zero_f32<<<64, 256, 0, stream>>>(henc[1][0], B_ * H_);
  zero_f32<<<64, 256, 0, stream>>>(hinf[0],    B_ * H_);
  zero_f16<<<64, 256, 0, stream>>>(henc16[0][0], B_ * H_);
  zero_f16<<<64, 256, 0, stream>>>(henc16[1][0], B_ * H_);
  zero_f16<<<64, 256, 0, stream>>>(hinf16[0],    B_ * H_);
  zero_f32<<<8,  256, 0, stream>>>(cbuf[0],    B_ * S_);

  // Bidirectional encoder: fwd + bwd fused per launch (blockIdx.y = dir).
  for (int s = 0; s < T_; ++s) {
    enc_step<<<dim3(32, 2, 1), 256, 0, stream>>>(
        x16, Pbf_ih, Pbf_hh, bf_bih, bf_bhh,
        Pbb_ih, Pbb_hh, bb_bih, bb_bhh,
        henc[0][s & 1], henc[0][(s + 1) & 1],
        henc16[0][s & 1], henc16[0][(s + 1) & 1],
        henc[1][s & 1], henc[1][(s + 1) & 1],
        henc16[1][s & 1], henc16[1][(s + 1) & 1],
        hb16, s);
  }

  // Inference scan: GRU GEMM step then fused heads.
  for (int t = 0; t < T_; ++t) {
    inf_step<<<dim3(32, 1, 1), 256, 0, stream>>>(
        hb16, cbuf[t & 1], hinf[t & 1], hinf[(t + 1) & 1],
        hinf16[t & 1], hinf16[(t + 1) & 1],
        Pin_ih, Pin_hh, in_bih, in_bhh, t);
    inf_small<<<dim3(B_, 1, 1), 256, 0, stream>>>(
        hinf[(t + 1) & 1], cbuf[t & 1], cbuf[(t + 1) & 1],
        x, gum, tau,
        fc_w, fc_b, tr_w1, tr_b1, tr_w2, tr_b2,
        em_w1, em_b1, em_w2, em_b2,
        kl_tb, rc_tb, out + 3, t);
  }

  final_reduce<<<1, 256, 0, stream>>>(kl_tb, rc_tb, out);
}
